// RelGraphConvN_1451698946528
// MI455X (gfx1250) — compile-verified, matrix operand order checked by reading, not run on hardware
//
#include <hip/hip_runtime.h>

// ---------------------------------------------------------------------------
// R-GCN (2 layers, basis decomposition) + mean_nodes pooling for MI455X
// Fused gather -> v_wmma_f32_16x16x32_bf16 -> atomic scatter.
// M-blocked (64 edges / wave); A-tiles staged via GLOBAL_LOAD_ASYNC_TO_LDS;
// group descriptors forced wave-uniform (readfirstlane) so the common-case
// scatter runs with full EXEC and no per-atomic exec save/restore.
// ---------------------------------------------------------------------------

typedef __bf16 bf16;
typedef __attribute__((ext_vector_type(16))) __bf16 v16bf;
typedef __attribute__((ext_vector_type(8)))  __bf16 v8bf;
typedef __attribute__((ext_vector_type(8)))  float  v8f;
typedef __attribute__((ext_vector_type(4)))  int    v4i;

#define N_NODES  100000
#define N_EDGES  1000000
#define N_GRAPHS 64
#define IN_F     64
#define H_F      64
#define N_CLS    16
#define N_RELS   8
#define MB       4            // M-tiles (of 16 edges) per wave in msg kernels

#if __has_builtin(__builtin_amdgcn_global_load_async_to_lds_b128) && \
    __has_builtin(__builtin_amdgcn_s_wait_asynccnt)
#define HAVE_ASYNC_LDS 1
typedef __attribute__((address_space(1))) v4i gv4i;   // global int4
typedef __attribute__((address_space(3))) v4i lv4i;   // LDS int4
#else
#define HAVE_ASYNC_LDS 0
#endif

// -------------------- workspace layout (bytes) ------------------------------
static constexpr size_t OFF_H1   = 0;            // 100000*64*4 = 25,600,000 (layer1 agg)
static constexpr size_t OFF_H2   = 25600000;     // 100000*16*4 =  6,400,000 (layer2 agg)
static constexpr size_t OFF_GSUM = 32000000;     // 64*16*4 = 4096
static constexpr size_t OFF_GCNT = 32004096;     // 64*4 = 256
static constexpr size_t OFF_BCNT = 32004352;     // 8*4 = 32
static constexpr size_t ZERO_U32 = 32004384 / 4;
static constexpr size_t OFF_META = 32004608;     // binOff[9]@+0 cursors[9]@+64 g64Pref[9]@+128
static constexpr size_t OFF_SORT = 32004864;     // 1,000,000 ints
static constexpr size_t OFF_XBF  = 36004864;     // 100000*64*2
static constexpr size_t OFF_HBF  = 48804864;     // 100000*64*2
static constexpr size_t OFF_W1F  = 61604864;     // 8*64*64*2 (B-frag packed)
static constexpr size_t OFF_W2F  = 61670400;     // 8*64*16*2
static constexpr size_t OFF_S1F  = 61686784;     // 64*64*2
static constexpr size_t OFF_S2F  = 61694976;     // 64*16*2

// ---------------------------------------------------------------------------
// CDNA5 16-bit B-fragment packing: (k,n) in a 32-wide k-step:
//   lane = n + (((k&15)>=8)?16:0) ; j = (k&7) + ((k>=16)?8:0)
// ---------------------------------------------------------------------------
__device__ __forceinline__ void frag_pos(int kk, int nn, int& lane, int& j) {
    lane = nn + (((kk & 15) >= 8) ? 16 : 0);
    j    = (kk & 7) + ((kk >= 16) ? 8 : 0);
}

// A-fragment: lane owns row (L&15); K0=(L<16?0:8); halves [K0..K0+7],[K0+16..K0+23].
__device__ __forceinline__ v16bf load_a(const bf16* row, int kt, int K0) {
    const bf16* p = row + kt * 32 + K0;
    v8bf lo = *(const v8bf*)(p);
    v8bf hi = *(const v8bf*)(p + 16);
    return __builtin_shufflevector(lo, hi, 0,1,2,3,4,5,6,7,8,9,10,11,12,13,14,15);
}

__device__ __forceinline__ v8f wmma_bf16(v16bf a, v16bf b, v8f c) {
    return __builtin_amdgcn_wmma_f32_16x16x32_bf16(false, a, false, b, (short)0, c, false, false);
}

#if HAVE_ASYNC_LDS
__device__ __forceinline__ void async_gather_b128(const bf16* g, void* l) {
    __builtin_amdgcn_global_load_async_to_lds_b128((gv4i*)g, (lv4i*)l, 0, 0);
}
#endif

// -------------------- prep kernels ------------------------------------------

__global__ void zero_ws(unsigned* __restrict__ p, size_t n) {
    for (size_t i = blockIdx.x * (size_t)blockDim.x + threadIdx.x; i < n;
         i += (size_t)gridDim.x * blockDim.x) p[i] = 0u;
}

__global__ void to_bf16(const float* __restrict__ x, bf16* __restrict__ y, int n) {
    for (int i = blockIdx.x * blockDim.x + threadIdx.x; i < n; i += gridDim.x * blockDim.x)
        y[i] = (bf16)x[i];
}

__global__ void prep_weights(const float* __restrict__ bases1, const float* __restrict__ coeff1,
                             const float* __restrict__ bases2, const float* __restrict__ coeff2,
                             const float* __restrict__ wself1, const float* __restrict__ wself2,
                             bf16* __restrict__ W1f, bf16* __restrict__ W2f,
                             bf16* __restrict__ S1f, bf16* __restrict__ S2f) {
    const int total = 32768 + 8192 + 4096 + 1024;
    for (int id = blockIdx.x * blockDim.x + threadIdx.x; id < total;
         id += gridDim.x * blockDim.x) {
        if (id < 32768) {                       // W1 = coeff1 x bases1 : [8][64k][64n]
            int r = id >> 12, k = (id >> 6) & 63, n = id & 63;
            float v = 0.f;
            for (int b = 0; b < 8; ++b) v += coeff1[r * 8 + b] * bases1[(b * 64 + k) * 64 + n];
            int kt = k >> 5, kk = k & 31, nt = n >> 4, nn = n & 15, lane, j;
            frag_pos(kk, nn, lane, j);
            W1f[(size_t)((((r * 2 + kt) * 4 + nt) * 32 + lane) << 4) + j] = (bf16)v;
        } else if (id < 32768 + 8192) {         // W2 : [8][64k][16n]
            int t = id - 32768;
            int r = t >> 10, k = (t >> 4) & 63, n = t & 15;
            float v = 0.f;
            for (int b = 0; b < 8; ++b) v += coeff2[r * 8 + b] * bases2[(b * 64 + k) * 16 + n];
            int kt = k >> 5, kk = k & 31, lane, j;
            frag_pos(kk, n, lane, j);
            W2f[(size_t)(((r * 2 + kt) * 32 + lane) << 4) + j] = (bf16)v;
        } else if (id < 32768 + 8192 + 4096) {  // wself1 : [64k][64n]
            int t = id - 40960;
            int k = t >> 6, n = t & 63;
            int kt = k >> 5, kk = k & 31, nt = n >> 4, nn = n & 15, lane, j;
            frag_pos(kk, nn, lane, j);
            S1f[(size_t)(((kt * 4 + nt) * 32 + lane) << 4) + j] = (bf16)wself1[k * 64 + n];
        } else {                                // wself2 : [64k][16n]
            int t = id - 45056;
            int k = t >> 4, n = t & 15;
            int kt = k >> 5, kk = k & 31, lane, j;
            frag_pos(kk, n, lane, j);
            S2f[(size_t)((kt * 32 + lane) << 4) + j] = (bf16)wself2[k * 16 + n];
        }
    }
}

// -------------------- edge binning by relation ------------------------------

__global__ void edge_hist(const int* __restrict__ ety, unsigned* __restrict__ bcnt) {
    for (int e = blockIdx.x * blockDim.x + threadIdx.x; e < N_EDGES; e += gridDim.x * blockDim.x)
        atomicAdd(&bcnt[ety[e]], 1u);
}

__global__ void edge_prefix(const unsigned* __restrict__ bcnt, int* __restrict__ binOff,
                            int* __restrict__ cursors, int* __restrict__ g64Pref) {
    int off = 0, tp = 0;
    binOff[0] = 0; g64Pref[0] = 0;
    for (int r = 0; r < N_RELS; ++r) {
        cursors[r] = off;
        off += (int)bcnt[r];
        binOff[r + 1] = off;
        tp += ((int)bcnt[r] + 63) >> 6;       // 64-edge groups per relation
        g64Pref[r + 1] = tp;
    }
}

__global__ void edge_scatter(const int* __restrict__ ety, int* __restrict__ cursors,
                             int* __restrict__ sorted) {
    for (int e = blockIdx.x * blockDim.x + threadIdx.x; e < N_EDGES; e += gridDim.x * blockDim.x) {
        int pos = atomicAdd(&cursors[ety[e]], 1);
        sorted[pos] = e;
    }
}

// -------------------- layer 1: edge messages (64 -> 64) ---------------------
// One wave per 64-edge single-relation group. Group descriptors are forced
// wave-uniform so r/ebase/cntg live in SGPRs; full groups (the common case)
// scatter with full EXEC and no per-atomic exec manipulation.
__global__ void l1_msg(const bf16* __restrict__ xbf, const int* __restrict__ src,
                       const int* __restrict__ dst, const int* __restrict__ sorted,
                       const int* __restrict__ binOff, const int* __restrict__ g64Pref,
                       const bf16* __restrict__ W1f, float* __restrict__ h1) {
    int lane = threadIdx.x & 31;
    int wid  = threadIdx.x >> 5;
    int g = __builtin_amdgcn_readfirstlane(blockIdx.x * (blockDim.x >> 5) + wid);
    if (g >= g64Pref[N_RELS]) return;
    int r = 0;
    while (g >= g64Pref[r + 1]) ++r;            // scalar loop
    int ebase = binOff[r] + (g - g64Pref[r]) * 64;
    int cntg = binOff[r + 1] - ebase; cntg = cntg < 64 ? cntg : 64;
    cntg = __builtin_amdgcn_readfirstlane(cntg);
    bool full = (cntg == 64);                   // scalar branch condition

    __builtin_prefetch(sorted + ebase + 256, 0, 1);   // global_prefetch_b8

    int i = lane & 15;
    int sN[MB], dN[MB];
#pragma unroll
    for (int mt = 0; mt < MB; ++mt) {
        int idx = mt * 16 + i; idx = idx < cntg ? idx : cntg - 1;
        int e = sorted[ebase + idx];
        sN[mt] = src[e];
        dN[mt] = dst[e];
    }

#if HAVE_ASYNC_LDS
    __shared__ __align__(16) char smem[4][MB * 2048];
    char* my = smem[wid];
#pragma unroll
    for (int mt = 0; mt < MB; ++mt) {
#pragma unroll
        for (int q = 0; q < 4; ++q) {
            int chunk = q * 32 + lane;          // 128 x 16B chunks per 16x128B tile
            int row = chunk >> 3, piece = chunk & 7;
            int srow = __shfl(sN[mt], row);
            async_gather_b128(xbf + (size_t)srow * 64 + piece * 8,
                              my + mt * 2048 + chunk * 16);
        }
    }
    __builtin_amdgcn_s_wait_asynccnt(0);
#endif

    int K0 = (lane < 16) ? 0 : 8;
    int Mb = (lane < 16) ? 0 : 8, nn = lane & 15;
    v8f vz = {};

    for (int nt = 0; nt < 4; ++nt) {            // n-tile outermost: low VGPR pressure
        v8f C[MB];
#pragma unroll
        for (int mt = 0; mt < MB; ++mt) C[mt] = vz;

#pragma unroll
        for (int kt = 0; kt < 2; ++kt) {
            v16bf b = *(const v16bf*)(W1f + (((size_t)((r * 2 + kt) * 4 + nt) * 32 + lane) << 4));
#pragma unroll
            for (int mt = 0; mt < MB; ++mt) {
#if HAVE_ASYNC_LDS
                v16bf a = load_a((const bf16*)(my + mt * 2048 + i * 128), kt, K0);  // ds_load_b128
#else
                v16bf a = load_a(xbf + (size_t)sN[mt] * 64, kt, K0);
#endif
                C[mt] = wmma_bf16(a, b, C[mt]);
            }
        }
        if (full) {                             // scalar branch, full-EXEC scatter
#pragma unroll
            for (int mt = 0; mt < MB; ++mt) {
#pragma unroll
                for (int v = 0; v < 8; ++v) {
                    int dm = __shfl(dN[mt], Mb + v);   // ds_bpermute row->dst
                    atomicAdd(&h1[(size_t)dm * 64 + nt * 16 + nn], C[mt][v]);
                }
            }
        } else {                                // tail group: per-lane guard
#pragma unroll
            for (int mt = 0; mt < MB; ++mt) {
                int cmt = cntg - mt * 16; cmt = cmt < 0 ? 0 : (cmt > 16 ? 16 : cmt);
#pragma unroll
                for (int v = 0; v < 8; ++v) {
                    int m  = Mb + v;
                    int dm = __shfl(dN[mt], m);
                    if (m < cmt)
                        atomicAdd(&h1[(size_t)dm * 64 + nt * 16 + nn], C[mt][v]);
                }
            }
        }
    }
}

// layer 1 finish: h = relu(agg + x @ wself1 + bias1); emit bf16 for layer 2.
__global__ void l1_finish(const bf16* __restrict__ xbf, const float* __restrict__ bias1,
                          const bf16* __restrict__ S1f, const float* __restrict__ h1,
                          bf16* __restrict__ hbf) {
    int lane = threadIdx.x & 31;
    int tile = blockIdx.x * (blockDim.x >> 5) + (threadIdx.x >> 5);
    if (tile >= N_NODES / 16) return;
    int nb = tile * 16;
    int K0 = (lane < 16) ? 0 : 8;
    int Mb = (lane < 16) ? 0 : 8, nn = lane & 15;
    const bf16* row = xbf + (size_t)(nb + (lane & 15)) * 64;

    v8f c0 = {}, c1 = {}, c2 = {}, c3 = {};
#pragma unroll
    for (int v = 0; v < 8; ++v) {               // seed C with agg + bias
        size_t o = (size_t)(nb + Mb + v) * 64 + nn;
        c0[v] = h1[o]      + bias1[nn];
        c1[v] = h1[o + 16] + bias1[16 + nn];
        c2[v] = h1[o + 32] + bias1[32 + nn];
        c3[v] = h1[o + 48] + bias1[48 + nn];
    }
#pragma unroll
    for (int kt = 0; kt < 2; ++kt) {
        v16bf a = load_a(row, kt, K0);
        const bf16* bp = S1f + (((size_t)kt * 4 * 32 + lane) << 4);
        c0 = wmma_bf16(a, *(const v16bf*)(bp),           c0);
        c1 = wmma_bf16(a, *(const v16bf*)(bp + 1 * 512), c1);
        c2 = wmma_bf16(a, *(const v16bf*)(bp + 2 * 512), c2);
        c3 = wmma_bf16(a, *(const v16bf*)(bp + 3 * 512), c3);
    }
#pragma unroll
    for (int v = 0; v < 8; ++v) {
        size_t o = (size_t)(nb + Mb + v) * 64 + nn;
        hbf[o]      = (bf16)fmaxf(c0[v], 0.f);
        hbf[o + 16] = (bf16)fmaxf(c1[v], 0.f);
        hbf[o + 32] = (bf16)fmaxf(c2[v], 0.f);
        hbf[o + 48] = (bf16)fmaxf(c3[v], 0.f);
    }
}

// -------------------- layer 2: edge messages (64 -> 16) ---------------------
__global__ void l2_msg(const bf16* __restrict__ hbf, const int* __restrict__ src,
                       const int* __restrict__ dst, const int* __restrict__ sorted,
                       const int* __restrict__ binOff, const int* __restrict__ g64Pref,
                       const bf16* __restrict__ W2f, float* __restrict__ h2) {
    int lane = threadIdx.x & 31;
    int wid  = threadIdx.x >> 5;
    int g = __builtin_amdgcn_readfirstlane(blockIdx.x * (blockDim.x >> 5) + wid);
    if (g >= g64Pref[N_RELS]) return;
    int r = 0;
    while (g >= g64Pref[r + 1]) ++r;
    int ebase = binOff[r] + (g - g64Pref[r]) * 64;
    int cntg = binOff[r + 1] - ebase; cntg = cntg < 64 ? cntg : 64;
    cntg = __builtin_amdgcn_readfirstlane(cntg);
    bool full = (cntg == 64);

    int i = lane & 15;
    int sN[MB], dN[MB];
#pragma unroll
    for (int mt = 0; mt < MB; ++mt) {
        int idx = mt * 16 + i; idx = idx < cntg ? idx : cntg - 1;
        int e = sorted[ebase + idx];
        sN[mt] = src[e];
        dN[mt] = dst[e];
    }

#if HAVE_ASYNC_LDS
    __shared__ __align__(16) char smem[4][MB * 2048];
    char* my = smem[wid];
#pragma unroll
    for (int mt = 0; mt < MB; ++mt) {
#pragma unroll
        for (int q = 0; q < 4; ++q) {
            int chunk = q * 32 + lane;
            int row = chunk >> 3, piece = chunk & 7;
            int srow = __shfl(sN[mt], row);
            async_gather_b128(hbf + (size_t)srow * 64 + piece * 8,
                              my + mt * 2048 + chunk * 16);
        }
    }
    __builtin_amdgcn_s_wait_asynccnt(0);
#endif

    int K0 = (lane < 16) ? 0 : 8;
    v8f vz = {};
    v8f C[MB];
#pragma unroll
    for (int mt = 0; mt < MB; ++mt) C[mt] = vz;

#pragma unroll
    for (int kt = 0; kt < 2; ++kt) {
        v16bf b = *(const v16bf*)(W2f + (((size_t)(r * 2 + kt) * 32 + lane) << 4));
#pragma unroll
        for (int mt = 0; mt < MB; ++mt) {
#if HAVE_ASYNC_LDS
            v16bf a = load_a((const bf16*)(my + mt * 2048 + i * 128), kt, K0);
#else
            v16bf a = load_a(hbf + (size_t)sN[mt] * 64, kt, K0);
#endif
            C[mt] = wmma_bf16(a, b, C[mt]);
        }
    }

    int Mb = (lane < 16) ? 0 : 8, nn = lane & 15;
    if (full) {                                 // scalar branch, full-EXEC scatter
#pragma unroll
        for (int mt = 0; mt < MB; ++mt) {
#pragma unroll
            for (int v = 0; v < 8; ++v) {
                int dm = __shfl(dN[mt], Mb + v);
                atomicAdd(&h2[(size_t)dm * 16 + nn], C[mt][v]);
            }
        }
    } else {
#pragma unroll
        for (int mt = 0; mt < MB; ++mt) {
            int cmt = cntg - mt * 16; cmt = cmt < 0 ? 0 : (cmt > 16 ? 16 : cmt);
#pragma unroll
            for (int v = 0; v < 8; ++v) {
                int m  = Mb + v;
                int dm = __shfl(dN[mt], m);
                if (m < cmt) atomicAdd(&h2[(size_t)dm * 16 + nn], C[mt][v]);
            }
        }
    }
}

// layer 2 finish: relu(agg + h @ wself2 + bias2), accumulate per-graph sums.
__global__ void l2_finish(const bf16* __restrict__ hbf, const float* __restrict__ bias2,
                          const bf16* __restrict__ S2f, const float* __restrict__ h2,
                          const int* __restrict__ gid, float* __restrict__ gsum) {
    int lane = threadIdx.x & 31;
    int tile = blockIdx.x * (blockDim.x >> 5) + (threadIdx.x >> 5);
    if (tile >= N_NODES / 16) return;
    int nb = tile * 16;
    int K0 = (lane < 16) ? 0 : 8;
    int Mb = (lane < 16) ? 0 : 8, nn = lane & 15;
    const bf16* row = hbf + (size_t)(nb + (lane & 15)) * 64;

    v8f c = {};
#pragma unroll
    for (int v = 0; v < 8; ++v)
        c[v] = h2[(size_t)(nb + Mb + v) * 16 + nn] + bias2[nn];
#pragma unroll
    for (int kt = 0; kt < 2; ++kt) {
        v16bf a = load_a(row, kt, K0);
        v16bf b = *(const v16bf*)(S2f + (((size_t)kt * 32 + lane) << 4));
        c = wmma_bf16(a, b, c);
    }
#pragma unroll
    for (int v = 0; v < 8; ++v) {
        int node = nb + Mb + v;
        atomicAdd(&gsum[(size_t)gid[node] * 16 + nn], fmaxf(c[v], 0.f));
    }
}

__global__ void node_counts(const int* __restrict__ gid, float* __restrict__ gcnt) {
    for (int n = blockIdx.x * blockDim.x + threadIdx.x; n < N_NODES; n += gridDim.x * blockDim.x)
        atomicAdd(&gcnt[gid[n]], 1.0f);
}

__global__ void finalize(const float* __restrict__ gsum, const float* __restrict__ gcnt,
                         float* __restrict__ out) {
    int i = blockIdx.x * blockDim.x + threadIdx.x;
    if (i < N_GRAPHS * N_CLS) out[i] = gsum[i] / fmaxf(gcnt[i >> 4], 1.0f);
}

// -------------------- host launcher -----------------------------------------

extern "C" void kernel_launch(void* const* d_in, const int* in_sizes, int n_in,
                              void* d_out, int out_size, void* d_ws, size_t ws_size,
                              hipStream_t stream) {
    const float* in_feat = (const float*)d_in[0];
    const int*   src     = (const int*)d_in[1];
    const int*   dst     = (const int*)d_in[2];
    const int*   ety     = (const int*)d_in[3];
    const int*   gid     = (const int*)d_in[4];
    const float* bases1  = (const float*)d_in[5];
    const float* coeff1  = (const float*)d_in[6];
    const float* wself1  = (const float*)d_in[7];
    const float* bias1   = (const float*)d_in[8];
    const float* bases2  = (const float*)d_in[9];
    const float* coeff2  = (const float*)d_in[10];
    const float* wself2  = (const float*)d_in[11];
    const float* bias2   = (const float*)d_in[12];
    float*       out     = (float*)d_out;

    char* ws = (char*)d_ws;
    float* h1      = (float*)(ws + OFF_H1);
    float* h2      = (float*)(ws + OFF_H2);
    float* gsum    = (float*)(ws + OFF_GSUM);
    float* gcnt    = (float*)(ws + OFF_GCNT);
    unsigned* bcnt = (unsigned*)(ws + OFF_BCNT);
    int* binOff    = (int*)(ws + OFF_META);
    int* cursors   = (int*)(ws + OFF_META + 64);
    int* g64Pref   = (int*)(ws + OFF_META + 128);
    int* sorted    = (int*)(ws + OFF_SORT);
    bf16* xbf      = (bf16*)(ws + OFF_XBF);
    bf16* hbf      = (bf16*)(ws + OFF_HBF);
    bf16* W1f      = (bf16*)(ws + OFF_W1F);
    bf16* W2f      = (bf16*)(ws + OFF_W2F);
    bf16* S1f      = (bf16*)(ws + OFF_S1F);
    bf16* S2f      = (bf16*)(ws + OFF_S2F);

    // Upper bound on single-relation 64-edge groups: E/64 + R = 15633.
    const int maxG       = N_EDGES / 64 + N_RELS;
    const int msgBlocks  = (maxG + 3) / 4;            // 4 waves/block (wave32)
    const int nodeTiles  = N_NODES / 16;              // 6250
    const int nodeBlocks = (nodeTiles + 3) / 4;

    zero_ws     <<<2048, 256, 0, stream>>>((unsigned*)ws, ZERO_U32);
    prep_weights<<<180, 256, 0, stream>>>(bases1, coeff1, bases2, coeff2,
                                          wself1, wself2, W1f, W2f, S1f, S2f);
    to_bf16     <<<4096, 256, 0, stream>>>(in_feat, xbf, N_NODES * IN_F);
    edge_hist   <<<2048, 256, 0, stream>>>(ety, bcnt);
    edge_prefix <<<1, 1, 0, stream>>>(bcnt, binOff, cursors, g64Pref);
    edge_scatter<<<2048, 256, 0, stream>>>(ety, cursors, sorted);

    l1_msg   <<<msgBlocks, 128, 0, stream>>>(xbf, src, dst, sorted, binOff, g64Pref, W1f, h1);
    l1_finish<<<nodeBlocks, 128, 0, stream>>>(xbf, bias1, S1f, h1, hbf);
    l2_msg   <<<msgBlocks, 128, 0, stream>>>(hbf, src, dst, sorted, binOff, g64Pref, W2f, h2);
    l2_finish<<<nodeBlocks, 128, 0, stream>>>(hbf, bias2, S2f, h2, gid, gsum);

    node_counts<<<512, 256, 0, stream>>>(gid, gcnt);
    finalize   <<<4, 256, 0, stream>>>(gsum, gcnt, out);

    (void)in_sizes; (void)n_in; (void)out_size; (void)ws_size;
}
